// TriangleAttention_49744311222640
// MI455X (gfx1250) — matrix operand order repaired and checked by reading, not compiled, not run on hardware
//
#include <hip/hip_runtime.h>

// ---------------------------------------------------------------------------
// TriangleAttention for MI455X (gfx1250): wave32 + v_wmma_f32_16x16x32_f16
// + GLOBAL_LOAD_ASYNC_TO_LDS_B128 staging (ASYNCcnt) for LDS tiles.
// S=256, H=4, DH=32, D=128
// ---------------------------------------------------------------------------

typedef __attribute__((ext_vector_type(16))) _Float16 v16h;
typedef __attribute__((ext_vector_type(8)))  _Float16 v8h;
typedef __attribute__((ext_vector_type(4)))  _Float16 v4h;
typedef __attribute__((ext_vector_type(8)))  float    v8f;

#define S_DIM 256
#define H_DIM 4
#define DH_DIM 32
#define D_DIM 128
#define M_TOT (S_DIM * S_DIM)          // 65536 rows

__device__ __forceinline__ v16h make_v16(v8h lo, v8h hi) {
  v16h r;
#pragma unroll
  for (int i = 0; i < 8; ++i) { r[i] = lo[i]; r[i + 8] = hi[i]; }
  return r;
}

// A fragment, 16x32 f16, row-major source: elem A[m][k] at base[m*ld + k].
// lane (m = lane&15, kh = lane>>4); k segments at kh*8 and 16+kh*8 (16B each).
__device__ __forceinline__ v16h frag_a_rm(const _Float16* base, int ld) {
  int lane = threadIdx.x & 31;
  int m = lane & 15, kh = lane >> 4;
  const _Float16* p = base + m * ld + kh * 8;
  v8h lo = *(const v8h*)(p);
  v8h hi = *(const v8h*)(p + 16);
  return make_v16(lo, hi);
}

// B fragment, 32x16 f16, column-major source: elem B[k][n] at base[n*ld + k].
// lane (n = lane&15, kh = lane>>4); k segments at kh*8 and 16+kh*8 (16B each).
__device__ __forceinline__ v16h frag_b_cm(const _Float16* base, int ld) {
  int lane = threadIdx.x & 31;
  int n = lane & 15, kh = lane >> 4;
  const _Float16* p = base + n * ld + kh * 8;
  v8h lo = *(const v8h*)(p);
  v8h hi = *(const v8h*)(p + 16);
  return make_v16(lo, hi);
}

__device__ __forceinline__ v8f wmma_f16(v16h a, v16h b, v8f c) {
  return __builtin_amdgcn_wmma_f32_16x16x32_f16(false, a, false, b,
                                                (short)0, c, false, false);
}

// Async global->LDS 16B copy (GVS mode: saddr64 + vaddr32). Tracked ASYNCcnt.
__device__ __forceinline__ void async_b128(unsigned int lds_off,
                                           const void* base,
                                           unsigned int voff) {
  asm volatile("global_load_async_to_lds_b128 %0, %1, %2"
               :
               : "v"(lds_off), "v"(voff),
                 "s"((unsigned long long)(uintptr_t)base)
               : "memory");
}
__device__ __forceinline__ void wait_async0() {
  asm volatile("s_wait_asynccnt 0x0" ::: "memory");
}
// Low 32 bits of a generic LDS address == LDS byte offset (aperture rules).
__device__ __forceinline__ unsigned int lds_off32(const void* p) {
  return (unsigned int)(uintptr_t)p;
}

// ---------------------------------------------------------------------------
// Kernel 0: convert 5 weight matrices (128x128 f32) -> f16, packed.
// ---------------------------------------------------------------------------
__global__ __launch_bounds__(256) void wcvt_kernel(
    const float* __restrict__ Wq, const float* __restrict__ Wk,
    const float* __restrict__ Wv, const float* __restrict__ Wg,
    const float* __restrict__ Wo, _Float16* __restrict__ out) {
  int idx = blockIdx.x * 256 + threadIdx.x;  // 0..81919
  int m = idx >> 14, off = idx & 16383;
  const float* src = (m == 0) ? Wq : (m == 1) ? Wk : (m == 2) ? Wv
                                                : (m == 3) ? Wg : Wo;
  out[idx] = (_Float16)src[off];
}

// ---------------------------------------------------------------------------
// Kernel 1: LayerNorm fp32 -> f16, one wave per row of 128
// ---------------------------------------------------------------------------
__global__ __launch_bounds__(256) void ln_kernel(
    const float* __restrict__ x, const float* __restrict__ lw,
    const float* __restrict__ lb, _Float16* __restrict__ xn) {
  int w = threadIdx.x >> 5, lane = threadIdx.x & 31;
  size_t row = (size_t)blockIdx.x * 8 + w;
  const float* xr = x + row * D_DIM + lane * 4;
  float4 v = *(const float4*)xr;
  float s  = v.x + v.y + v.z + v.w;
  float ss = v.x * v.x + v.y * v.y + v.z * v.z + v.w * v.w;
#pragma unroll
  for (int m = 1; m < 32; m <<= 1) {
    s  += __shfl_xor(s, m);
    ss += __shfl_xor(ss, m);
  }
  float mu   = s * (1.0f / 128.0f);
  float var  = ss * (1.0f / 128.0f) - mu * mu;
  float rstd = rsqrtf(var + 1e-5f);
  float4 wv = *(const float4*)(lw + lane * 4);
  float4 bv = *(const float4*)(lb + lane * 4);
  v4h o;
  o[0] = (_Float16)((v.x - mu) * rstd * wv.x + bv.x);
  o[1] = (_Float16)((v.y - mu) * rstd * wv.y + bv.y);
  o[2] = (_Float16)((v.z - mu) * rstd * wv.z + bv.z);
  o[3] = (_Float16)((v.w - mu) * rstd * wv.w + bv.w);
  *(v4h*)(xn + row * D_DIM + lane * 4) = o;
}

// ---------------------------------------------------------------------------
// Kernel 2: C[m][n] = sum_k A[m][k] * W[n][k]  (A f16 [M,128], W f16 [128,128])
// mode 0: f16 out, scatter to [b][h][q][dh]   (Q/K/V)
// mode 1: f16 out, sigmoid applied, row-major  (gate)
// mode 2: f32 out, row-major                   (final output)
// Block = 256 thr (8 waves): 64(M) x 128(N) tile; wave -> 16(M) x 64(N).
// A and W tiles staged with async global->LDS.
// ---------------------------------------------------------------------------
__global__ __launch_bounds__(256) void gemm_xw_kernel(
    const _Float16* __restrict__ A, const _Float16* __restrict__ W,
    void* __restrict__ out, int mode) {
  __shared__ alignas(16) _Float16 Ash[64][136];   // 272B rows
  __shared__ alignas(16) _Float16 Wsh[128][136];
  const int tid = threadIdx.x;
  const int m_base = blockIdx.x * 64;

  { // async stage W: 128 rows x 16 chunks of 16B = 2048 chunks, 8/thread
    unsigned int wdst = lds_off32(&Wsh[0][0]);
#pragma unroll
    for (int c = 0; c < 8; ++c) {
      int chunk = tid * 8 + c;
      int row = chunk >> 4, seg = chunk & 15;
      async_b128(wdst + row * 272 + seg * 16, W, row * 256 + seg * 16);
    }
    // async stage A tile: 64 rows x 16 chunks = 1024 chunks, 4/thread
    unsigned int adst = lds_off32(&Ash[0][0]);
    const _Float16* abase = A + (size_t)m_base * D_DIM;
#pragma unroll
    for (int c = 0; c < 4; ++c) {
      int chunk = tid * 4 + c;
      int row = chunk >> 4, seg = chunk & 15;
      async_b128(adst + row * 272 + seg * 16, abase, row * 256 + seg * 16);
    }
    wait_async0();
  }
  __syncthreads();

  const int w = tid >> 5, lane = tid & 31, kh = lane >> 4, nl = lane & 15;
  const int m0 = (w >> 1) * 16;
  const int n0 = (w & 1) * 64;
  v8f acc[4] = {v8f{}, v8f{}, v8f{}, v8f{}};
#pragma unroll
  for (int kc = 0; kc < 4; ++kc) {
    v16h aF = frag_a_rm(&Ash[m0][kc * 32], 136);
#pragma unroll
    for (int t = 0; t < 4; ++t) {
      v16h bF = frag_b_cm(&Wsh[n0 + t * 16][kc * 32], 136);
      acc[t] = wmma_f16(aF, bF, acc[t]);
    }
  }
#pragma unroll
  for (int t = 0; t < 4; ++t) {
#pragma unroll
    for (int r = 0; r < 8; ++r) {
      int m = m0 + r + 8 * kh;
      int n = n0 + t * 16 + nl;
      size_t gm = (size_t)m_base + m;
      float v = acc[t][r];
      if (mode == 0) {
        int b = (int)(gm >> 8), q = (int)(gm & 255);
        int h = n >> 5, dh = n & 31;
        ((_Float16*)out)[(((size_t)b * H_DIM + h) * S_DIM + q) * DH_DIM + dh] =
            (_Float16)v;
      } else if (mode == 1) {
        ((_Float16*)out)[gm * D_DIM + n] =
            (_Float16)(1.0f / (1.0f + __expf(-v)));
      } else {
        ((float*)out)[gm * D_DIM + n] = v;
      }
    }
  }
}

// ---------------------------------------------------------------------------
// Kernel 3: biasM[h][q][k] = dot(xn[q,k,:], Wb[h,:]) + mask[q,k]
// Block = 256 thr: 128(M) x 16(N) tile (only n<4 stored).
// ---------------------------------------------------------------------------
__global__ __launch_bounds__(256) void bias_gemm_kernel(
    const _Float16* __restrict__ A, const float* __restrict__ Wb,
    const float* __restrict__ mask, float* __restrict__ biasM) {
  __shared__ alignas(16) _Float16 Ash[128][136];
  __shared__ alignas(16) _Float16 Wsh[16][136];
  const int tid = threadIdx.x;
  const int m_base = blockIdx.x * 128;

  { // stage Wb (4x128), zero-pad rows 4..15
    int idx = tid * 8;
    int n = idx >> 7, k = idx & 127;
#pragma unroll
    for (int i = 0; i < 8; ++i)
      Wsh[n][k + i] = (n < H_DIM) ? (_Float16)Wb[n * D_DIM + k + i]
                                  : (_Float16)0.0f;
  }
  { // async stage A tile: 128 rows x 16 chunks = 2048 chunks, 8/thread
    unsigned int adst = lds_off32(&Ash[0][0]);
    const _Float16* abase = A + (size_t)m_base * D_DIM;
#pragma unroll
    for (int c = 0; c < 8; ++c) {
      int chunk = tid * 8 + c;
      int row = chunk >> 4, seg = chunk & 15;
      async_b128(adst + row * 272 + seg * 16, abase, row * 256 + seg * 16);
    }
    wait_async0();
  }
  __syncthreads();

  const int w = tid >> 5, lane = tid & 31, kh = lane >> 4, nl = lane & 15;
  const int m0 = w * 16;
  v8f acc = {};
#pragma unroll
  for (int kc = 0; kc < 4; ++kc) {
    v16h aF = frag_a_rm(&Ash[m0][kc * 32], 136);
    v16h bF = frag_b_cm(&Wsh[0][kc * 32], 136);
    acc = wmma_f16(aF, bF, acc);
  }
  if (nl < H_DIM) {
#pragma unroll
    for (int r = 0; r < 8; ++r) {
      size_t gm = (size_t)m_base + m0 + r + 8 * kh;
      biasM[(size_t)nl * M_TOT + gm] = acc[r] + mask[gm];
    }
  }
}

// ---------------------------------------------------------------------------
// Kernel 4: flash attention per (b,h) block + gating epilogue.
// K async-staged row-major, V staged transposed in LDS.
// Probs go through per-wave LDS buffer for C-layout -> A-layout conversion.
// ---------------------------------------------------------------------------
__global__ __launch_bounds__(256) void attn_kernel(
    const _Float16* __restrict__ Qb, const _Float16* __restrict__ Kb,
    const _Float16* __restrict__ Vb, const float* __restrict__ biasM,
    const _Float16* __restrict__ SG, _Float16* __restrict__ Afin) {
  __shared__ alignas(16) _Float16 Ksh[256][40];    // 80B rows
  __shared__ alignas(16) _Float16 Vt[32][264];     // 528B rows
  __shared__ alignas(16) _Float16 pbuf[8][16][40];
  const int tid = threadIdx.x;
  const int bh = blockIdx.x;            // b*H + h
  const int b = bh >> 2, h = bh & 3;
  const size_t base = (size_t)bh * S_DIM * DH_DIM;

  { // async stage K: 256 rows x 4 chunks of 16B = 1024 chunks, 4/thread
    unsigned int kdst = lds_off32(&Ksh[0][0]);
    const _Float16* kg = Kb + base;
#pragma unroll
    for (int c = 0; c < 4; ++c) {
      int chunk = tid * 4 + c;
      int row = chunk >> 2, seg = chunk & 3;
      async_b128(kdst + row * 80 + seg * 16, kg, row * 64 + seg * 16);
    }
  }
  { // V -> LDS transposed (through VGPRs; transpose in flight)
    const _Float16* vr = Vb + base + (size_t)tid * DH_DIM;
    v8h t0 = *(const v8h*)(vr + 0),  t1 = *(const v8h*)(vr + 8);
    v8h t2 = *(const v8h*)(vr + 16), t3 = *(const v8h*)(vr + 24);
#pragma unroll
    for (int d = 0; d < 8; ++d) {
      Vt[d][tid]      = t0[d];
      Vt[d + 8][tid]  = t1[d];
      Vt[d + 16][tid] = t2[d];
      Vt[d + 24][tid] = t3[d];
    }
  }
  wait_async0();
  __syncthreads();

  const int w = tid >> 5, lane = tid & 31, kh = lane >> 4, nl = lane & 15;
  const float scale = 0.17677669529663687f;  // 1/sqrt(32)
  const float* bmr = biasM + (size_t)h * M_TOT;

  for (int pass = 0; pass < 2; ++pass) {
    const int q0 = (pass * 8 + w) * 16;
    v16h aq = frag_a_rm(Qb + base + (size_t)q0 * DH_DIM, DH_DIM);
    v8f o0 = {}, o1 = {};
    float mrun[8], lrun[8];
#pragma unroll
    for (int r = 0; r < 8; ++r) { mrun[r] = -1e30f; lrun[r] = 0.0f; }

    for (int kc = 0; kc < 8; ++kc) {
      const int key0 = kc * 32;
      v16h bk0 = frag_b_cm(&Ksh[key0][0], 40);
      v16h bk1 = frag_b_cm(&Ksh[key0 + 16][0], 40);
      v8f z = {};
      v8f s0 = wmma_f16(aq, bk0, z);
      v8f s1 = wmma_f16(aq, bk1, z);
#pragma unroll
      for (int r = 0; r < 8; ++r) {
        int m = r + 8 * kh;
        size_t rowoff = (size_t)(q0 + m) * S_DIM + key0;
        float x0 = s0[r] * scale + bmr[rowoff + nl];
        float x1 = s1[r] * scale + bmr[rowoff + 16 + nl];
        float vmax = fmaxf(x0, x1);
        vmax = fmaxf(vmax, __shfl_xor(vmax, 1));
        vmax = fmaxf(vmax, __shfl_xor(vmax, 2));
        vmax = fmaxf(vmax, __shfl_xor(vmax, 4));
        vmax = fmaxf(vmax, __shfl_xor(vmax, 8));
        float mnew = fmaxf(mrun[r], vmax);
        float corr = __expf(mrun[r] - mnew);
        float p0 = __expf(x0 - mnew);
        float p1 = __expf(x1 - mnew);
        float rs = p0 + p1;
        rs += __shfl_xor(rs, 1);
        rs += __shfl_xor(rs, 2);
        rs += __shfl_xor(rs, 4);
        rs += __shfl_xor(rs, 8);
        lrun[r] = lrun[r] * corr + rs;
        mrun[r] = mnew;
        o0[r] *= corr;
        o1[r] *= corr;
        pbuf[w][m][nl]      = (_Float16)p0;   // C-layout -> LDS
        pbuf[w][m][16 + nl] = (_Float16)p1;
      }
      v16h pF  = frag_a_rm(&pbuf[w][0][0], 40);   // LDS -> A-layout
      v16h bv0 = frag_b_cm(&Vt[0][key0], 264);
      v16h bv1 = frag_b_cm(&Vt[16][key0], 264);
      o0 = wmma_f16(pF, bv0, o0);
      o1 = wmma_f16(pF, bv1, o1);
    }

    // normalize + gate: Afin[b*256+q][h*32+dh] = o/l * sigmoid(g)
#pragma unroll
    for (int r = 0; r < 8; ++r) {
      int m = r + 8 * kh;
      float inv = 1.0f / fmaxf(lrun[r], 1e-30f);
      size_t grow = (size_t)b * S_DIM + (q0 + m);
      int c0col = h * DH_DIM + nl, c1col = c0col + 16;
      float g0 = (float)SG[grow * D_DIM + c0col];
      float g1 = (float)SG[grow * D_DIM + c1col];
      Afin[grow * D_DIM + c0col] = (_Float16)(o0[r] * inv * g0);
      Afin[grow * D_DIM + c1col] = (_Float16)(o1[r] * inv * g1);
    }
  }
}

// ---------------------------------------------------------------------------
extern "C" void kernel_launch(void* const* d_in, const int* in_sizes, int n_in,
                              void* d_out, int out_size, void* d_ws,
                              size_t ws_size, hipStream_t stream) {
  (void)in_sizes; (void)n_in; (void)out_size; (void)ws_size;
  const float* x    = (const float*)d_in[0];
  const float* mask = (const float*)d_in[1];
  const float* ln_w = (const float*)d_in[2];
  const float* ln_b = (const float*)d_in[3];
  const float* Wb   = (const float*)d_in[4];
  const float* Wq   = (const float*)d_in[5];
  const float* Wk   = (const float*)d_in[6];
  const float* Wv   = (const float*)d_in[7];
  const float* Wg   = (const float*)d_in[8];
  const float* Wo   = (const float*)d_in[9];

  char* ws = (char*)d_ws;
  const size_t CH = (size_t)M_TOT * D_DIM * sizeof(_Float16);  // 16 MiB
  _Float16* XN   = (_Float16*)(ws);            // xn, later reused as gated-o
  _Float16* Q    = (_Float16*)(ws + 1 * CH);   // [b][h][q][dh]
  _Float16* K    = (_Float16*)(ws + 2 * CH);
  _Float16* V    = (_Float16*)(ws + 3 * CH);
  _Float16* SG   = (_Float16*)(ws + 4 * CH);   // sigmoid(g), row-major
  float*    BIAS = (float*)   (ws + 5 * CH);   // [H][S][S] with mask folded
  _Float16* WH   = (_Float16*)(ws + 5 * CH + (1u << 20));  // 5x16384 f16

  _Float16* WqH = WH + 0 * 16384;
  _Float16* WkH = WH + 1 * 16384;
  _Float16* WvH = WH + 2 * 16384;
  _Float16* WgH = WH + 3 * 16384;
  _Float16* WoH = WH + 4 * 16384;

  wcvt_kernel<<<320, 256, 0, stream>>>(Wq, Wk, Wv, Wg, Wo, WH);
  ln_kernel<<<M_TOT / 8, 256, 0, stream>>>(x, ln_w, ln_b, XN);
  gemm_xw_kernel<<<M_TOT / 64, 256, 0, stream>>>(XN, WqH, Q, 0);
  gemm_xw_kernel<<<M_TOT / 64, 256, 0, stream>>>(XN, WkH, K, 0);
  gemm_xw_kernel<<<M_TOT / 64, 256, 0, stream>>>(XN, WvH, V, 0);
  gemm_xw_kernel<<<M_TOT / 64, 256, 0, stream>>>(XN, WgH, SG, 1);
  bias_gemm_kernel<<<M_TOT / 128, 256, 0, stream>>>(XN, Wb, mask, BIAS);
  attn_kernel<<<S_DIM * H_DIM, 256, 0, stream>>>(Q, K, V, BIAS, SG, XN);
  gemm_xw_kernel<<<M_TOT / 64, 256, 0, stream>>>(XN, WoH, d_out, 2);
}
